// CrossAttention_12128987643944
// MI455X (gfx1250) — matrix operand order repaired and checked by reading, not compiled, not run on hardware
//
#include <hip/hip_runtime.h>
#include <hip/hip_bf16.h>

#define NATOM   65536
#define LMAXP   160
#define DIMD    128
#define SCL     0.17677669529663687f   /* 32^-0.5 */

typedef __attribute__((ext_vector_type(16))) __bf16        v16bf;
typedef __attribute__((ext_vector_type(8)))  float         v8f;
typedef __attribute__((ext_vector_type(4)))  float         f4v;
typedef __attribute__((ext_vector_type(4)))  unsigned int  u4v;
typedef __attribute__((ext_vector_type(2)))  unsigned int  u2v;

__device__ __forceinline__ unsigned short f2bf(float f) {
  unsigned u = __float_as_uint(f);
  u = (u + 0x7FFFu + ((u >> 16) & 1u)) >> 16;
  return (unsigned short)u;
}

__device__ __forceinline__ v16bf frag_pair(u4v lo, u4v hi) {
  v16bf f;
  *(u4v*)&f       = lo;
  *((u4v*)&f + 1) = hi;
  return f;
}
// A-fragment: 16 contiguous halfs per lane
__device__ __forceinline__ v16bf ldFragA(const unsigned short* p) {
  return frag_pair(*(const u4v*)p, *(const u4v*)(p + 8));
}
// B-fragment: 16 contiguous halfs per lane
__device__ __forceinline__ v16bf ldFragB(const unsigned short* p) {
  return frag_pair(*(const u4v*)p, *(const u4v*)(p + 8));
}

__device__ __forceinline__ v8f wmma_bf16(v16bf a, v16bf b, v8f c) {
  return __builtin_amdgcn_wmma_f32_16x16x32_bf16(false, a, false, b, (short)0, c,
                                                 false, false);
}

__device__ __forceinline__ int lower_bound_i32(const int* a, int n, int v) {
  int lo = 0, hi = n;
  while (lo < hi) { int m = (lo + hi) >> 1; if (a[m] < v) lo = m + 1; else hi = m; }
  return lo;
}

// ---- Kernel 1: swizzle the 4 weight matrices into WMMA B-fragment order (bf16) ----
// Layout: ws[w*16384 + nt*2048 + c*512 + lane*16 + e], w in {Wq,Wk,Wv,Wo}
__global__ void prep_weights(const float* __restrict__ Wq, const float* __restrict__ Wk,
                             const float* __restrict__ Wv, const float* __restrict__ Wo,
                             unsigned short* __restrict__ wsw) {
  int tid  = blockIdx.x * 256 + threadIdx.x;          // 65536 elements total
  int w    = tid >> 14;
  int rem  = tid & 16383;
  int nt   = rem >> 11;
  int c    = (rem & 2047) >> 9;
  int le   = rem & 511;
  int lane = le >> 4, e = le & 15;
  int n    = lane & 15, khi = lane >> 4;
  int k    = c * 32 + khi * 16 + e;
  const float* W = (w == 0) ? Wq : (w == 1) ? Wk : (w == 2) ? Wv : Wo;
  wsw[tid] = f2bf(W[k * DIMD + nt * 16 + n]);
}

// ---- Kernel 2: fully fused pad + QKV proj + masked attention + out proj + LN ----
__global__ __launch_bounds__(256) void attn_kernel(
    const float* __restrict__ x_don, const float* __restrict__ x_acc,
    const unsigned short* __restrict__ wsw,
    const float* __restrict__ bq, const float* __restrict__ bk,
    const float* __restrict__ bv, const float* __restrict__ bo,
    const float* __restrict__ ln_g, const float* __restrict__ ln_b,
    const int* __restrict__ batch_don, const int* __restrict__ batch_acc,
    float* __restrict__ out) {
  extern __shared__ unsigned char smem_raw[];
  const int dir  = blockIdx.x & 1;
  const int pair = blockIdx.x >> 1;
  const float* xq    = dir ? x_acc : x_don;
  const float* xkv   = dir ? x_don : x_acc;
  const int*   barrq = dir ? batch_acc : batch_don;
  const int*   barrk = dir ? batch_don : batch_acc;

  const int sq = lower_bound_i32(barrq, NATOM, pair);
  const int eq = lower_bound_i32(barrq, NATOM, pair + 1);
  const int sk = lower_bound_i32(barrk, NATOM, pair);
  const int ek = lower_bound_i32(barrk, NATOM, pair + 1);
  const int cq_raw = eq - sq;
  const int cq = cq_raw < LMAXP ? cq_raw : LMAXP;
  const int ck = (ek - sk) < LMAXP ? (ek - sk) : LMAXP;
  if (cq <= 0) return;
  const int nqt = (cq + 15) >> 4;
  const int nkt = (ck + 15) >> 4;
  const int kcN = (nkt + 1) >> 1;

  // LDS regions (bytes): Qb 40960 | Kb 40960 | Vt 49152 | D 90112  = 221184
  unsigned short* Qb   = (unsigned short*)smem_raw;      // [160][128] bf16
  unsigned short* Kb   = Qb + LMAXP * DIMD;              // [160][128] bf16
  unsigned short* Vt   = Kb + LMAXP * DIMD;              // [128][192] bf16 (transposed V)
  unsigned short* Dh   = Vt + DIMD * 192;                // overlapped region
  unsigned short* Xq   = Dh;                             // phase1: [160][128]
  unsigned short* Xkv  = Dh + LMAXP * DIMD;              // phase1: [160][128]
  unsigned short* Slab = Dh;                             // phase2: 8 waves x [16][192]
  unsigned short* Ab   = Dh + 8 * 16 * 192;              // [160][128] attended bf16
  float*          Obuf = (float*)smem_raw;               // phase3: [160][128] f32 (over Qb+Kb)

  const int tid  = threadIdx.x;
  const int lane = tid & 31, wave = tid >> 5;
  const int ln   = lane & 15, lhi = lane >> 4;

  // ---------- Phase 1a: stage x rows to LDS as bf16 (zero-padded to tile) ----------
  for (int i = tid; i < nqt * 16 * 32; i += 256) {
    int row = i >> 5, c4 = i & 31;
    f4v v = {0.f, 0.f, 0.f, 0.f};
    if (row < cq) v = *(const f4v*)(xq + (size_t)(sq + row) * DIMD + c4 * 4);
    u2v p;
    p.x = (unsigned)f2bf(v.x) | ((unsigned)f2bf(v.y) << 16);
    p.y = (unsigned)f2bf(v.z) | ((unsigned)f2bf(v.w) << 16);
    *(u2v*)(Xq + row * DIMD + c4 * 4) = p;
  }
  for (int i = tid; i < nkt * 16 * 32; i += 256) {
    int row = i >> 5, c4 = i & 31;
    f4v v = {0.f, 0.f, 0.f, 0.f};
    if (row < ck) v = *(const f4v*)(xkv + (size_t)(sk + row) * DIMD + c4 * 4);
    u2v p;
    p.x = (unsigned)f2bf(v.x) | ((unsigned)f2bf(v.y) << 16);
    p.y = (unsigned)f2bf(v.z) | ((unsigned)f2bf(v.w) << 16);
    *(u2v*)(Xkv + row * DIMD + c4 * 4) = p;
  }
  __syncthreads();

  // ---------- Phase 1b: Q/K/V projections via WMMA ----------
  {
    const int Tq = nqt * 8, Tk = nkt * 8, Tt = Tq + 2 * Tk;
    for (int t = wave; t < Tt; t += 8) {
      int proj, u;
      if (t < Tq)            { proj = 0; u = t; }
      else if (t < Tq + Tk)  { proj = 1; u = t - Tq; }
      else                   { proj = 2; u = t - Tq - Tk; }
      const int mt = u >> 3, nt = u & 7;
      const unsigned short* X  = (proj == 0) ? Xq : Xkv;
      const unsigned short* wb = wsw + proj * 16384;
      v8f acc = {0.f, 0.f, 0.f, 0.f, 0.f, 0.f, 0.f, 0.f};
#pragma unroll
      for (int c = 0; c < 4; ++c) {
        v16bf a = ldFragA(X + (mt * 16 + ln) * DIMD + c * 32 + 16 * lhi);
        v16bf b = ldFragB(wb + (nt * 4 + c) * 512 + lane * 16);
        acc = wmma_bf16(a, b, acc);
      }
      const float* bias = (proj == 0) ? bq : (proj == 1) ? bk : bv;
      const float bn = bias[nt * 16 + ln];
      if (proj < 2) {
        unsigned short* Y = (proj == 0) ? Qb : Kb;
#pragma unroll
        for (int v = 0; v < 8; ++v)
          Y[(mt * 16 + v + 8 * lhi) * DIMD + nt * 16 + ln] = f2bf(acc[v] + bn);
      } else {  // V stored transposed [hd][key], 8 contiguous halfs -> b128 store
        u4v pk;
        pk.x = (unsigned)f2bf(acc[0] + bn) | ((unsigned)f2bf(acc[1] + bn) << 16);
        pk.y = (unsigned)f2bf(acc[2] + bn) | ((unsigned)f2bf(acc[3] + bn) << 16);
        pk.z = (unsigned)f2bf(acc[4] + bn) | ((unsigned)f2bf(acc[5] + bn) << 16);
        pk.w = (unsigned)f2bf(acc[6] + bn) | ((unsigned)f2bf(acc[7] + bn) << 16);
        *(u4v*)(Vt + (nt * 16 + ln) * 192 + mt * 16 + 8 * lhi) = pk;
      }
    }
  }
  __syncthreads();

  // ---------- Phase 2: per-head attention (8 waves = 4 heads x 2 q-halves) ----------
  {
    const int h = wave >> 1, s = wave & 1;
    unsigned short* slab = Slab + wave * 16 * 192;
    for (int qt = s; qt < nqt; qt += 2) {
      v16bf aQ = ldFragA(Qb + (qt * 16 + ln) * DIMD + h * 32 + 16 * lhi);
      v8f sacc[10];
#pragma unroll
      for (int kt = 0; kt < 10; ++kt) {
        v8f z = {0.f, 0.f, 0.f, 0.f, 0.f, 0.f, 0.f, 0.f};
        if (kt < nkt) {
          v16bf bK = ldFragB(Kb + (kt * 16 + ln) * DIMD + h * 32 + 16 * lhi);
          sacc[kt] = wmma_bf16(aQ, bK, z);
        } else {
          sacc[kt] = z;
        }
      }
      // masked softmax, row-wise over the 16-lane half groups
#pragma unroll
      for (int v = 0; v < 8; ++v) {
        float m = -1e30f;
#pragma unroll
        for (int kt = 0; kt < 10; ++kt) {
          if (kt < nkt) {
            int cg = kt * 16 + ln;
            float x = (cg < ck) ? sacc[kt][v] * SCL : -1e30f;
            sacc[kt][v] = x;
            m = fmaxf(m, x);
          }
        }
#pragma unroll
        for (int o = 1; o < 16; o <<= 1) m = fmaxf(m, __shfl_xor(m, o, 32));
        float sum = 0.f;
#pragma unroll
        for (int kt = 0; kt < 10; ++kt) {
          if (kt < nkt) {
            int cg = kt * 16 + ln;
            float e = (cg < ck) ? __expf(sacc[kt][v] - m) : 0.f;
            sacc[kt][v] = e;
            sum += e;
          }
        }
#pragma unroll
        for (int o = 1; o < 16; o <<= 1) sum += __shfl_xor(sum, o, 32);
        const float rinv = 1.f / fmaxf(sum, 1e-9f);
#pragma unroll
        for (int kt = 0; kt < 10; ++kt)
          if (kt < nkt)
            slab[(v + 8 * lhi) * 192 + kt * 16 + ln] = f2bf(sacc[kt][v] * rinv);
      }
      // zero-pad probability columns up to the 32-wide k-chunk boundary
      for (int col = nkt * 16 + lane; col < kcN * 32; col += 32)
#pragma unroll
        for (int r = 0; r < 16; ++r) slab[r * 192 + col] = 0;

      // attended = attn @ V  (V transposed in LDS -> contiguous B-fragments)
#pragma unroll
      for (int nh = 0; nh < 2; ++nh) {
        v8f acc = {0.f, 0.f, 0.f, 0.f, 0.f, 0.f, 0.f, 0.f};
        for (int kc = 0; kc < kcN; ++kc) {
          v16bf a = ldFragA(slab + ln * 192 + kc * 32 + 16 * lhi);
          v16bf b = ldFragB(Vt + (h * 32 + nh * 16 + ln) * 192 + kc * 32 + 16 * lhi);
          acc = wmma_bf16(a, b, acc);
        }
#pragma unroll
        for (int v = 0; v < 8; ++v)
          Ab[(qt * 16 + v + 8 * lhi) * DIMD + h * 32 + nh * 16 + ln] = f2bf(acc[v]);
      }
    }
  }
  __syncthreads();

  // ---------- Phase 3: output projection (attended @ Wo + bo) ----------
  for (int t = wave; t < nqt * 8; t += 8) {
    const int mt = t >> 3, nt = t & 7;
    v8f acc = {0.f, 0.f, 0.f, 0.f, 0.f, 0.f, 0.f, 0.f};
#pragma unroll
    for (int c = 0; c < 4; ++c) {
      v16bf a = ldFragA(Ab + (mt * 16 + ln) * DIMD + c * 32 + 16 * lhi);
      v16bf b = ldFragB(wsw + 3 * 16384 + (nt * 4 + c) * 512 + lane * 16);
      acc = wmma_bf16(a, b, acc);
    }
    const float bn = bo[nt * 16 + ln];
#pragma unroll
    for (int v = 0; v < 8; ++v)
      Obuf[(mt * 16 + v + 8 * lhi) * DIMD + nt * 16 + ln] = acc[v] + bn;
  }
  __syncthreads();

  // ---------- Phase 4: residual + LayerNorm + contiguous scatter to out ----------
  float* outp = out + (size_t)(dir ? NATOM : 0) * DIMD;
  for (int r = tid; r < cq_raw; r += 256) {
    const int rr = (r < cq - 1) ? r : (cq - 1);   // JAX-style clip for overflow rows
    const float* xr = xq + (size_t)(sq + rr) * DIMD;
    const float* orow = Obuf + rr * DIMD;
    float sum = 0.f, ss = 0.f;
#pragma unroll 4
    for (int i = 0; i < DIMD; i += 4) {
      f4v a = *(const f4v*)(orow + i);
      f4v b = *(const f4v*)(xr + i);
      float y0 = a.x + b.x, y1 = a.y + b.y, y2 = a.z + b.z, y3 = a.w + b.w;
      sum += y0 + y1 + y2 + y3;
      ss  += y0 * y0 + y1 * y1 + y2 * y2 + y3 * y3;
    }
    const float mu = sum * (1.f / DIMD);
    const float var = ss * (1.f / DIMD) - mu * mu;
    const float rstd = rsqrtf(var + 1e-5f);
    float* dst = outp + (size_t)(sq + r) * DIMD;
#pragma unroll 4
    for (int i = 0; i < DIMD; i += 4) {
      f4v a = *(const f4v*)(orow + i);
      f4v b = *(const f4v*)(xr + i);
      f4v g = *(const f4v*)(ln_g + i);
      f4v bb = *(const f4v*)(ln_b + i);
      f4v rv;
      rv.x = (a.x + b.x - mu) * rstd * g.x + bb.x;
      rv.y = (a.y + b.y - mu) * rstd * g.y + bb.y;
      rv.z = (a.z + b.z - mu) * rstd * g.z + bb.z;
      rv.w = (a.w + b.w - mu) * rstd * g.w + bb.w;
      *(f4v*)(dst + i) = rv;
    }
  }
}

extern "C" void kernel_launch(void* const* d_in, const int* in_sizes, int n_in,
                              void* d_out, int out_size, void* d_ws, size_t ws_size,
                              hipStream_t stream) {
  (void)in_sizes; (void)n_in; (void)out_size; (void)ws_size;
  const float* x_don = (const float*)d_in[0];
  const float* x_acc = (const float*)d_in[1];
  const float* Wq = (const float*)d_in[2];
  const float* bq = (const float*)d_in[3];
  const float* Wk = (const float*)d_in[4];
  const float* bk = (const float*)d_in[5];
  const float* Wv = (const float*)d_in[6];
  const float* bv = (const float*)d_in[7];
  const float* Wo = (const float*)d_in[8];
  const float* bo = (const float*)d_in[9];
  const float* ln_g = (const float*)d_in[10];
  const float* ln_b = (const float*)d_in[11];
  const int* batch_don = (const int*)d_in[12];
  const int* batch_acc = (const int*)d_in[13];
  float* out = (float*)d_out;
  unsigned short* wsw = (unsigned short*)d_ws;

  prep_weights<<<256, 256, 0, stream>>>(Wq, Wk, Wv, Wo, wsw);

  const int smem_bytes = 221184;  // 40960 + 40960 + 49152 + 90112
  hipFuncSetAttribute((const void*)attn_kernel,
                      hipFuncAttributeMaxDynamicSharedMemorySize, smem_bytes);
  attn_kernel<<<2048, 256, smem_bytes, stream>>>(
      x_don, x_acc, wsw, bq, bk, bv, bo, ln_g, ln_b, batch_don, batch_acc, out);
}
